// NormLinearAttention_83700322665173
// MI455X (gfx1250) — compile-verified, matrix operand order checked by reading, not run on hardware
//
#include <hip/hip_runtime.h>

// ---------------------------------------------------------------------------
// MiniMax-style lightning attention for MI455X (gfx1250, wave32).
// All GEMMs: v_wmma_f32_16x16x32_f16, operands staged into LDS with
// double-buffered global_load_async_to_lds_b128 (ASYNCcnt) DMA.
// ---------------------------------------------------------------------------

typedef __attribute__((ext_vector_type(16))) _Float16 v16h;
typedef __attribute__((ext_vector_type(8)))  _Float16 v8h;
typedef __attribute__((ext_vector_type(8)))  float    v8f;

#define LS  40            // LDS tile row stride in halves (32 + 8 pad)
#define LSB (128 * LS)    // halves per LDS tile buffer

// Async DMA one 16-byte packet from global memory into LDS (ASYNCcnt-tracked).
__device__ __forceinline__ void async_b128(unsigned lds_off, const void* g) {
  asm volatile("global_load_async_to_lds_b128 %0, %1, off"
               :: "v"(lds_off), "v"(g) : "memory");
}

// Load one 16x32 f16 operand fragment per CDNA5 WMMA VGPR layout:
// lane holds row (lane&15); elems 0..7 -> K = khi*8+0..7, elems 8..15 -> +16.
__device__ __forceinline__ v16h frag(const _Float16* rowp, int khi) {
  v8h lo = *(const v8h*)(rowp + (khi << 3));
  v8h hi = *(const v8h*)(rowp + 16 + (khi << 3));
  return __builtin_shufflevector(lo, hi, 0,1,2,3,4,5,6,7,8,9,10,11,12,13,14,15);
}

// Workgroup GEMM core: C[128x128] += A * B^T over K, both operands f16 in
// global memory, contraction-contiguous: A row-major [M][K], B row-major [N][K].
// 8 waves (wave32), each wave 64x32; double-buffered async LDS staging.
__device__ void wg_gemm16(v8f acc[4][2], const _Float16* Ab, int lda,
                          const _Float16* Bb, int ldb, int K,
                          _Float16* lA, _Float16* lB) {
  const int tid  = threadIdx.x;
  const int lane = tid & 31, w = tid >> 5;
  const int wm = w >> 2, wn = w & 3;
  const int l15 = lane & 15, khi = lane >> 4;
  const int jj = (tid & 3) << 3;     // half offset in row: 0,8,16,24
  const int i0 = tid >> 2;           // row 0..63 (and +64)
  const unsigned la0 = (unsigned)(size_t)lA;
  const unsigned lb0 = (unsigned)(size_t)lB;

  auto issue = [&](int k0, int buf) {   // 4 async b128 per thread / per wave
    const _Float16* ga = Ab + (size_t)i0 * lda + k0 + jj;
    const _Float16* gb = Bb + (size_t)i0 * ldb + k0 + jj;
    const unsigned lo = (unsigned)(buf * LSB + i0 * LS + jj) * 2u;
    async_b128(la0 + lo, ga);
    async_b128(lb0 + lo, gb);
    async_b128(la0 + lo + 64u * LS * 2u, ga + (size_t)64 * lda);
    async_b128(lb0 + lo + 64u * LS * 2u, gb + (size_t)64 * ldb);
  };

  issue(0, 0);
  const int nk = K >> 5;
  for (int ik = 0; ik < nk; ++ik) {
    const int cur = ik & 1;
    if (ik + 1 < nk) {
      issue((ik + 1) << 5, cur ^ 1);               // prefetch next tile
      asm volatile("s_wait_asynccnt 4" ::: "memory");  // tile ik done (in-order)
    } else {
      asm volatile("s_wait_asynccnt 0" ::: "memory");
    }
    __syncthreads();                                // all waves' DMA visible
    const _Float16* bA = lA + cur * LSB;
    const _Float16* bB = lB + cur * LSB;
    v16h b0 = frag(bB + (wn * 32 +      l15) * LS, khi);
    v16h b1 = frag(bB + (wn * 32 + 16 + l15) * LS, khi);
#pragma unroll
    for (int tm = 0; tm < 4; ++tm) {
      v16h a = frag(bA + (wm * 64 + tm * 16 + l15) * LS, khi);
      acc[tm][0] = __builtin_amdgcn_wmma_f32_16x16x32_f16(
          false, a, false, b0, (short)0, acc[tm][0], false, false);
      acc[tm][1] = __builtin_amdgcn_wmma_f32_16x16x32_f16(
          false, a, false, b1, (short)0, acc[tm][1], false, false);
    }
    __syncthreads();                                // readers done before overwrite
  }
}

// Iterate accumulator elements; f(tm, tn, r, mi, ni) with tile-local mi/ni.
template <typename F>
__device__ __forceinline__ void foreach_elem(F f) {
  const int lane = threadIdx.x & 31, w = threadIdx.x >> 5;
  const int wm = w >> 2, wn = w & 3, l15 = lane & 15, khi = lane >> 4;
#pragma unroll
  for (int tm = 0; tm < 4; ++tm)
#pragma unroll
    for (int tn = 0; tn < 2; ++tn)
#pragma unroll
      for (int r = 0; r < 8; ++r)
        f(tm, tn, r, wm * 64 + tm * 16 + khi * 8 + r, wn * 32 + tn * 16 + l15);
}

// ------------------------- f16 prep kernels ---------------------------------
__global__ void k_cvt16(const float* __restrict__ in, _Float16* __restrict__ out,
                        int total) {
  const int i = blockIdx.x * 256 + threadIdx.x;
  if (i < total) out[i] = (_Float16)in[i];
}

// WT[n][k] = (f16) W[k][n]   (W row-major [Kd][N])
__global__ void k_wtrans(const float* __restrict__ W, _Float16* __restrict__ WT,
                         int N, int Kd, int total) {
  const int i = blockIdx.x * 256 + threadIdx.x;
  if (i >= total) return;
  const int n = i % N, k = i / N;
  WT[(size_t)n * Kd + k] = (_Float16)W[(size_t)k * N + n];
}

// ------------------------- projections: out = x @ W, as f32 [B,Hx,L,D] ------
__global__ __launch_bounds__(256) void k_proj(const _Float16* __restrict__ x,
                                              const _Float16* __restrict__ WT,
                                              float* __restrict__ out, int Hx) {
  __shared__ __attribute__((aligned(16))) _Float16 lA[2 * LSB];
  __shared__ __attribute__((aligned(16))) _Float16 lB[2 * LSB];
  v8f acc[4][2] = {};
  const int m0 = blockIdx.y * 128, n0 = blockIdx.x * 128;
  wg_gemm16(acc, x + (size_t)m0 * 2048, 2048, WT + (size_t)n0 * 2048, 2048, 2048,
            lA, lB);
  foreach_elem([&](int tm, int tn, int r, int mi, int ni) {
    const int row = m0 + mi, col = n0 + ni;
    const int b = row >> 11, l = row & 2047;
    const int h = col >> 8,  d = col & 255;
    out[(((size_t)(b * Hx + h) * 2048) + l) * 256 + d] = acc[tm][tn][r];
  });
}

// -------- depthwise causal conv (K=4) + SiLU, f32 in -> f16 out, [B,Hx,L,D] -
__global__ void k_conv_silu(const float* __restrict__ raw, const float* __restrict__ wgt,
                            const float* __restrict__ bias, _Float16* __restrict__ out,
                            int Hx, int total) {
  const int idx = blockIdx.x * 256 + threadIdx.x;
  if (idx >= total) return;
  const int d  = idx & 255;
  const int l  = (idx >> 8) & 2047;
  const int bh = idx >> 19;
  const int c  = (bh % Hx) * 256 + d;
  float a = bias[c];
  const float* base = raw + ((size_t)bh * 2048) * 256 + d;
#pragma unroll
  for (int t = 0; t < 4; ++t) {
    const int ls = l - 3 + t;
    if (ls >= 0) a += wgt[c * 4 + t] * base[(size_t)ls * 256];
  }
  out[idx] = (_Float16)(a / (1.0f + __expf(-a)));     // SiLU
}

// ------------------------- per-chunk transposes (f16 out) -------------------
// vT[b,hk, e, n] = v[b,hk, c*256+n, e]
__global__ void k_transpose_v(const float* __restrict__ v, _Float16* __restrict__ vT,
                              int c, int total) {
  const int idx = blockIdx.x * 256 + threadIdx.x;
  if (idx >= total) return;
  const int n = idx & 255, e = (idx >> 8) & 255, bhk = idx >> 16;
  vT[idx] = (_Float16)v[(((size_t)bhk * 2048) + c * 256 + n) * 256 + e];
}

// kdT[b,h, d, n] = k16[b,h>>1, c*256+n, d] * exp(-s_h * (255 - n))
__global__ void k_make_kdT(const _Float16* __restrict__ k, const float* __restrict__ slope,
                           _Float16* __restrict__ kdT, int c, int total) {
  const int idx = blockIdx.x * 256 + threadIdx.x;
  if (idx >= total) return;
  const int n = idx & 255, d = (idx >> 8) & 255, bh = idx >> 16;
  const int b = bh >> 4, h = bh & 15, hk = h >> 1;
  const float s = slope[h];
  const float kv = (float)k[(((size_t)(b * 8 + hk) * 2048) + c * 256 + n) * 256 + d];
  kdT[idx] = (_Float16)(kv * __expf(-s * (255.0f - (float)n)));
}

// ------------------------- sc = (q_c @ k_c^T) * diag_decay (f16 out) --------
__global__ __launch_bounds__(256) void k_sc(const _Float16* __restrict__ q,
                                            const _Float16* __restrict__ kbuf,
                                            const float* __restrict__ slope,
                                            _Float16* __restrict__ sc, int c) {
  __shared__ __attribute__((aligned(16))) _Float16 lA[2 * LSB];
  __shared__ __attribute__((aligned(16))) _Float16 lB[2 * LSB];
  v8f acc[4][2] = {};
  const int m0 = blockIdx.y * 128, n0 = blockIdx.x * 128, bh = blockIdx.z;
  const int b = bh >> 4, h = bh & 15, hk = h >> 1;
  const float s = slope[h];
  wg_gemm16(acc, q + (((size_t)bh * 2048) + c * 256 + m0) * 256, 256,
            kbuf + (((size_t)(b * 8 + hk) * 2048) + c * 256 + n0) * 256, 256,
            256, lA, lB);
  foreach_elem([&](int tm, int tn, int r, int mi, int ni) {
    const int mg = m0 + mi, ng = n0 + ni;            // chunk-local positions
    const float dd = (mg >= ng) ? __expf(-s * (float)(mg - ng)) : 0.0f;
    sc[((size_t)bh * 256 + mg) * 256 + ng] = (_Float16)(acc[tm][tn][r] * dd);
  });
}

// -------- o = q_decay[m] * (q_c @ kv^T) + sc @ v_c, f32 [B,L,H*D] -----------
__global__ __launch_bounds__(256) void k_o(const _Float16* __restrict__ q,
                                           const _Float16* __restrict__ kvT,
                                           const _Float16* __restrict__ sc,
                                           const _Float16* __restrict__ vT,
                                           const float* __restrict__ slope,
                                           float* __restrict__ o, int c) {
  __shared__ __attribute__((aligned(16))) _Float16 lA[2 * LSB];
  __shared__ __attribute__((aligned(16))) _Float16 lB[2 * LSB];
  const int m0 = blockIdx.y * 128, e0 = blockIdx.x * 128, bh = blockIdx.z;
  const int b = bh >> 4, h = bh & 15, hk = h >> 1;
  const float s = slope[h];
  v8f acc1[4][2] = {};   // inter-chunk: q @ kv (kv mirrored f16, [e][d])
  wg_gemm16(acc1, q + (((size_t)bh * 2048) + c * 256 + m0) * 256, 256,
            kvT + ((size_t)bh * 256 + e0) * 256, 256, 256, lA, lB);
  v8f acc2[4][2] = {};   // intra-chunk: sc @ v (v transposed f16, [e][n])
  wg_gemm16(acc2, sc + ((size_t)bh * 256 + m0) * 256, 256,
            vT + ((size_t)(b * 8 + hk) * 256 + e0) * 256, 256, 256, lA, lB);
  foreach_elem([&](int tm, int tn, int r, int mi, int ni) {
    const int mg = m0 + mi, ei = e0 + ni;
    const float qd  = __expf(-s * (float)(mg + 1));   // q_decay in f32
    const float val = acc1[tm][tn][r] * qd + acc2[tm][tn][r];
    const int l = c * 256 + mg;
    o[((size_t)(b * 2048 + l)) * 4096 + h * 256 + ei] = val;
  });
}

// -------- kv[e][d] = blk_decay * kv + v^T @ (k*k_decay); f32 state + f16 mirror
__global__ __launch_bounds__(256) void k_kvupd(const _Float16* __restrict__ vT,
                                               const _Float16* __restrict__ kdT,
                                               const float* __restrict__ slope,
                                               float* __restrict__ kvf,
                                               _Float16* __restrict__ kv16) {
  __shared__ __attribute__((aligned(16))) _Float16 lA[2 * LSB];
  __shared__ __attribute__((aligned(16))) _Float16 lB[2 * LSB];
  v8f acc[4][2] = {};
  const int e0 = blockIdx.y * 128, d0 = blockIdx.x * 128, bh = blockIdx.z;
  const int b = bh >> 4, h = bh & 15, hk = h >> 1;
  wg_gemm16(acc, vT + ((size_t)(b * 8 + hk) * 256 + e0) * 256, 256,
            kdT + ((size_t)bh * 256 + d0) * 256, 256, 256, lA, lB);
  const float bd = __expf(-slope[h] * 256.0f);
  foreach_elem([&](int tm, int tn, int r, int mi, int ni) {
    const size_t off = ((size_t)bh * 256 + (e0 + mi)) * 256 + (d0 + ni);
    const float nv = kvf[off] * bd + acc[tm][tn][r];
    kvf[off]  = nv;
    kv16[off] = (_Float16)nv;
  });
}

// ------------------------- misc ---------------------------------------------
__global__ void k_zero(float* __restrict__ p, int total) {
  const int i = blockIdx.x * 256 + threadIdx.x;
  if (i < total) p[i] = 0.0f;
}
__global__ void k_zero16(_Float16* __restrict__ p, int total) {
  const int i = blockIdx.x * 256 + threadIdx.x;
  if (i < total) p[i] = (_Float16)0.0f;
}

// SimpleRMSNorm over 4096 channels; f32 in -> f16 out (for the final GEMM).
__global__ __launch_bounds__(256) void k_rms(const float* __restrict__ o,
                                             _Float16* __restrict__ o16) {
  __shared__ float red[256];
  const float* row = o + (size_t)blockIdx.x * 4096;
  _Float16* row16 = o16 + (size_t)blockIdx.x * 4096;
  const int tid = threadIdx.x;
  float ss = 0.0f;
  for (int i = tid; i < 4096; i += 256) { const float x = row[i]; ss += x * x; }
  red[tid] = ss; __syncthreads();
  for (int st = 128; st > 0; st >>= 1) {
    if (tid < st) red[tid] += red[tid + st];
    __syncthreads();
  }
  const float scale = rsqrtf(red[0] * (1.0f / 4096.0f) + 1e-6f);
  for (int i = tid; i < 4096; i += 256) row16[i] = (_Float16)(row[i] * scale);
}

// ------------------------- final projection: out = o_n @ Wo -----------------
__global__ __launch_bounds__(256) void k_outproj(const _Float16* __restrict__ o,
                                                 const _Float16* __restrict__ WoT,
                                                 float* __restrict__ out) {
  __shared__ __attribute__((aligned(16))) _Float16 lA[2 * LSB];
  __shared__ __attribute__((aligned(16))) _Float16 lB[2 * LSB];
  v8f acc[4][2] = {};
  const int m0 = blockIdx.y * 128, n0 = blockIdx.x * 128;
  wg_gemm16(acc, o + (size_t)m0 * 4096, 4096, WoT + (size_t)n0 * 4096, 4096, 4096,
            lA, lB);
  foreach_elem([&](int tm, int tn, int r, int mi, int ni) {
    out[(size_t)(m0 + mi) * 2048 + (n0 + ni)] = acc[tm][tn][r];
  });
}

// ---------------------------------------------------------------------------
extern "C" void kernel_launch(void* const* d_in, const int* in_sizes, int n_in,
                              void* d_out, int out_size, void* d_ws, size_t ws_size,
                              hipStream_t stream) {
  (void)in_sizes; (void)n_in; (void)out_size; (void)ws_size;
  const float* x     = (const float*)d_in[0];
  const float* Wq    = (const float*)d_in[1];
  const float* Wk    = (const float*)d_in[2];
  const float* Wv    = (const float*)d_in[3];
  const float* Wo    = (const float*)d_in[4];
  const float* cqw   = (const float*)d_in[5];
  const float* cqb   = (const float*)d_in[6];
  const float* ckw   = (const float*)d_in[7];
  const float* ckb   = (const float*)d_in[8];
  const float* slope = (const float*)d_in[9];
  float* out = (float*)d_out;
  char*  wsb = (char*)d_ws;

  // Workspace layout (bytes). qraw region is reused as the f32 o buffer;
  // kraw region is reused for vT16/kdT16/sc16; x16+WqT16 region is reused
  // for the 33.5MB f16 normalized-o buffer (all stream-ordered).
  float*    qraw = (float*)(wsb + 0);              // 64MB  [B,16,L,D] -> obuf
  float*    kraw = (float*)(wsb + 67108864);       // 32MB  [B,8,L,D]
  float*    vbuf = (float*)(wsb + 100663296);      // 32MB  [B,8,L,D]
  float*    kvf  = (float*)(wsb + 134217728);      // 8MB   [B*16,256e,256d]
  _Float16* kv16 = (_Float16*)(wsb + 142606336);   // 4MB   f16 mirror
  _Float16* q16  = (_Float16*)(wsb + 146800640);   // 32MB  [B,16,L,D]
  _Float16* k16  = (_Float16*)(wsb + 180355072);   // 16MB  [B,8,L,D]
  _Float16* x16  = (_Float16*)(wsb + 197132288);   // 16MB  [4096,2048]
  _Float16* wqT  = (_Float16*)(wsb + 213909504);   // 16MB  [4096,2048]
  _Float16* wkT  = (_Float16*)(wsb + 230686720);   // 8MB   [2048,2048]
  _Float16* wvT  = (_Float16*)(wsb + 239075328);   // 8MB   [2048,2048]
  _Float16* woT  = (_Float16*)(wsb + 247463936);   // 16MB  [2048,4096]
  _Float16* vT16  = (_Float16*)(wsb + 67108864);             // 2MB in kraw
  _Float16* kdT16 = (_Float16*)(wsb + 67108864 + 2097152);   // 4MB in kraw
  _Float16* sc16  = (_Float16*)(wsb + 67108864 + 6291456);   // 4MB in kraw
  float*    obuf  = qraw;
  _Float16* o16   = x16;                           // spans x16+wqT (33.5MB)

  const dim3 blk(256);
  // f16 operand preparation.
  k_cvt16<<<8388608 / 256, blk, 0, stream>>>(x, x16, 8388608);
  k_wtrans<<<8388608 / 256, blk, 0, stream>>>(Wq, wqT, 4096, 2048, 8388608);
  k_wtrans<<<4194304 / 256, blk, 0, stream>>>(Wk, wkT, 2048, 2048, 4194304);
  k_wtrans<<<4194304 / 256, blk, 0, stream>>>(Wv, wvT, 2048, 2048, 4194304);
  k_wtrans<<<8388608 / 256, blk, 0, stream>>>(Wo, woT, 2048, 4096, 8388608);
  // Q/K/V projections.
  k_proj<<<dim3(32, 32), blk, 0, stream>>>(x16, wqT, qraw, 16);
  k_proj<<<dim3(16, 32), blk, 0, stream>>>(x16, wkT, kraw, 8);
  k_proj<<<dim3(16, 32), blk, 0, stream>>>(x16, wvT, vbuf, 8);
  // Depthwise causal conv + SiLU (f16 outputs).
  k_conv_silu<<<16777216 / 256, blk, 0, stream>>>(qraw, cqw, cqb, q16, 16, 16777216);
  k_conv_silu<<<8388608 / 256, blk, 0, stream>>>(kraw, ckw, ckb, k16, 8, 8388608);
  // kv state init (f32 state + f16 mirror).
  k_zero<<<2097152 / 256, blk, 0, stream>>>(kvf, 2097152);
  k_zero16<<<2097152 / 256, blk, 0, stream>>>(kv16, 2097152);
  // Chunked lightning attention (8 chunks of 256, sequential via stream order).
  for (int c = 0; c < 8; ++c) {
    k_transpose_v<<<1048576 / 256, blk, 0, stream>>>(vbuf, vT16, c, 1048576);
    k_make_kdT<<<2097152 / 256, blk, 0, stream>>>(k16, slope, kdT16, c, 2097152);
    k_sc<<<dim3(2, 2, 32), blk, 0, stream>>>(q16, k16, slope, sc16, c);
    k_o<<<dim3(2, 2, 32), blk, 0, stream>>>(q16, kv16, sc16, vT16, slope, obuf, c);
    k_kvupd<<<dim3(2, 2, 32), blk, 0, stream>>>(vT16, kdT16, slope, kvf, kv16);
  }
  // SimpleRMSNorm -> f16 activations.
  k_rms<<<4096, blk, 0, stream>>>(obuf, o16);
  // Output projection into d_out [B,L,HID].
  k_outproj<<<dim3(16, 32), blk, 0, stream>>>(o16, woT, out);
}
